// CausalAttNet_15556371546338
// MI455X (gfx1250) — matrix operand order repaired and checked by reading, not compiled, not run on hardware
//
#include <hip/hip_runtime.h>
#include <hip/hip_bf16.h>

typedef __attribute__((ext_vector_type(16))) _Float16 v16h;
typedef __attribute__((ext_vector_type(8)))  _Float16 v8h;
typedef __attribute__((ext_vector_type(8)))  float    v8f;

#define NNODES   32768     // B*NPG
#define NEDGES   131072    // B*EPG
#define NGRAPHS  256
#define EPG      512
#define NRES     256
#define DIN      128
#define DMID     512       // 2*D
#define DHID     256       // D
#define BN_EPS   1e-5f

// ---------------------------------------------------------------- utilities
__global__ void k_zero_f32(float* __restrict__ p, int n) {
    int t = blockIdx.x * blockDim.x + threadIdx.x;
    if (t < n) p[t] = 0.f;
}

// W (K x Nn fp32 row-major) -> Wt (Nn x K f16 row-major) == column-major f16
__global__ void k_transpose_f16(const float* __restrict__ W, _Float16* __restrict__ Wt,
                                int K, int Nn) {
    int t = blockIdx.x * blockDim.x + threadIdx.x;
    if (t >= K * Nn) return;
    int n = t % Nn, k = t / Nn;
    Wt[(size_t)n * K + k] = (_Float16)W[(size_t)k * Nn + n];
}

// agg[col[e]*D + d] += h[row[e]*D + d]
__global__ void k_scatter_add(float* __restrict__ agg, const float* __restrict__ h,
                              const int* __restrict__ ei, int E, int logD) {
    int t = blockIdx.x * blockDim.x + threadIdx.x;
    int total = E << logD;
    if (t >= total) return;
    int e = t >> logD, d = t & ((1 << logD) - 1);
    int r = ei[e];
    int c = ei[E + e];
    atomicAdd(&agg[((size_t)c << logD) + d], h[((size_t)r << logD) + d]);
}

// o[i] = f16(a[i] + b[i])
__global__ void k_sum_to_f16(const float* __restrict__ a, const float* __restrict__ b,
                             _Float16* __restrict__ o, int n) {
    int t = blockIdx.x * blockDim.x + threadIdx.x;
    if (t < n) o[t] = (_Float16)(a[t] + b[t]);
}

// ------------------------------------------------------------- WMMA GEMM
// C(MxNn, f32, row-major) = A(MxK, f16, row-major) @ Bt^T + bias
// Bt is (Nn x K) f16 row-major (i.e. B column-major). One wave -> 16x64 strip.
__global__ void k_gemm_wmma(const _Float16* __restrict__ A, const _Float16* __restrict__ Bt,
                            const float* __restrict__ bias, float* __restrict__ C,
                            int M, int Nn, int K) {
    int wave = threadIdx.x >> 5;
    int lane = threadIdx.x & 31;
    int gw   = blockIdx.x * (blockDim.x >> 5) + wave;
    int numMT  = M >> 4;       // 16-row tiles
    int numNTg = Nn >> 6;      // 64-col groups
    if (gw >= numMT * numNTg) return;          // wave-uniform, EXEC stays full for WMMA
    int tileM = gw % numMT;
    int tileN = gw / numMT;
    int m0 = tileM << 4, n0 = tileN << 6;
    int laneLo = lane & 15;
    bool hi = lane >= 16;

    const _Float16* Arow = A + (size_t)(m0 + laneLo) * K;
    v8f acc0 = {}, acc1 = {}, acc2 = {}, acc3 = {};

    for (int k0 = 0; k0 < K; k0 += 32) {
        // A fragment: lane<16 -> K {k0..k0+7, k0+16..k0+23}; lane>=16 -> +8
        int kb = k0 + (hi ? 8 : 0);
        v8h alo = *(const v8h*)(Arow + kb);
        v8h ahi = *(const v8h*)(Arow + kb + 16);
        v16h a;
        #pragma unroll
        for (int i = 0; i < 8; ++i) { a[i] = alo[i]; a[i + 8] = ahi[i]; }

        // B fragments: lane<16 -> K {k0..k0+7, k0+16..k0+23}; lane>=16 -> +8
        int kbb = k0 + (hi ? 8 : 0);
        #pragma unroll
        for (int t = 0; t < 4; ++t) {
            const _Float16* Bcol = Bt + (size_t)(n0 + (t << 4) + laneLo) * K + kbb;
            v8h blo = *(const v8h*)(Bcol);
            v8h bhi = *(const v8h*)(Bcol + 16);
            v16h b;
            #pragma unroll
            for (int i = 0; i < 8; ++i) { b[i] = blo[i]; b[i + 8] = bhi[i]; }
            if      (t == 0) acc0 = __builtin_amdgcn_wmma_f32_16x16x32_f16(false, a, false, b, (short)0, acc0, false, false);
            else if (t == 1) acc1 = __builtin_amdgcn_wmma_f32_16x16x32_f16(false, a, false, b, (short)0, acc1, false, false);
            else if (t == 2) acc2 = __builtin_amdgcn_wmma_f32_16x16x32_f16(false, a, false, b, (short)0, acc2, false, false);
            else             acc3 = __builtin_amdgcn_wmma_f32_16x16x32_f16(false, a, false, b, (short)0, acc3, false, false);
        }
    }

    // C/D layout: VGPR v -> row m0+v (+8 for lanes>=16), col = n-subtile base + laneLo
    #pragma unroll
    for (int t = 0; t < 4; ++t) {
        v8f acc = (t == 0) ? acc0 : (t == 1) ? acc1 : (t == 2) ? acc2 : acc3;
        int ncol = n0 + (t << 4) + laneLo;
        float bv = bias[ncol];
        #pragma unroll
        for (int v = 0; v < 8; ++v) {
            int r = m0 + v + (hi ? 8 : 0);
            C[(size_t)r * Nn + ncol] = acc[v] + bv;
        }
    }
}

// ---------------------------------------------------------- BatchNorm stats
// partial per-column sum / sumsq over 128-row slabs, accumulate with atomics
__global__ void k_col_stats(const float* __restrict__ Z, float* __restrict__ gsum,
                            float* __restrict__ gsq, int C) {
    int r0 = blockIdx.x * 128;
    for (int c = threadIdx.x; c < C; c += blockDim.x) {
        float s = 0.f, q = 0.f;
        for (int r = 0; r < 128; ++r) {
            float v = Z[(size_t)(r0 + r) * C + c];
            s += v; q += v * v;
        }
        atomicAdd(&gsum[c], s);
        atomicAdd(&gsq[c], q);
    }
}

__global__ void k_finalize_stats(const float* __restrict__ gsum, const float* __restrict__ gsq,
                                 float* __restrict__ mu, float* __restrict__ rsig,
                                 int C, float invN) {
    int c = blockIdx.x * blockDim.x + threadIdx.x;
    if (c >= C) return;
    float m = gsum[c] * invN;
    float var = gsq[c] * invN - m * m;
    mu[c] = m;
    rsig[c] = rsqrtf(var + BN_EPS);
}

// f16(relu(bn(z)))
__global__ void k_bn_relu_f16(const float* __restrict__ Z, const float* __restrict__ mu,
                              const float* __restrict__ rsig, const float* __restrict__ g,
                              const float* __restrict__ be, _Float16* __restrict__ O,
                              int total, int cMask) {
    int t = blockIdx.x * blockDim.x + threadIdx.x;
    if (t >= total) return;
    int c = t & cMask;
    float v = (Z[t] - mu[c]) * rsig[c] * g[c] + be[c];
    O[t] = (_Float16)(v > 0.f ? v : 0.f);
}

// post = relu(bn(h)) + vemb[c]   (fp32)
__global__ void k_bn_relu_addv(const float* __restrict__ Z, const float* __restrict__ mu,
                               const float* __restrict__ rsig, const float* __restrict__ g,
                               const float* __restrict__ be, const float* __restrict__ vemb,
                               float* __restrict__ O, int total, int cMask) {
    int t = blockIdx.x * blockDim.x + threadIdx.x;
    if (t >= total) return;
    int c = t & cMask;
    float v = (Z[t] - mu[c]) * rsig[c] * g[c] + be[c];
    O[t] = (v > 0.f ? v : 0.f) + vemb[c];
}

// xn = bn(h)   (fp32, no relu)
__global__ void k_bn_f32(const float* __restrict__ Z, const float* __restrict__ mu,
                         const float* __restrict__ rsig, const float* __restrict__ g,
                         const float* __restrict__ be, float* __restrict__ O,
                         int total, int cMask) {
    int t = blockIdx.x * blockDim.x + threadIdx.x;
    if (t >= total) return;
    int c = t & cMask;
    O[t] = (Z[t] - mu[c]) * rsig[c] * g[c] + be[c];
}

// ------------------------------------------------------ edge scoring / sort
// s1[n] = xn[n,:].Wl[0:256], s2[n] = xn[n,:].Wl[256:512]  (one wave per node)
__global__ void k_node_scores(const float* __restrict__ xn, const float* __restrict__ Wl,
                              float* __restrict__ s1, float* __restrict__ s2) {
    int lane = threadIdx.x & 31;
    int n = blockIdx.x * (blockDim.x >> 5) + (threadIdx.x >> 5);
    const float* xr = xn + (size_t)n * DHID;
    float p1 = 0.f, p2 = 0.f;
    for (int i = lane; i < DHID; i += 32) {
        float v = xr[i];
        p1 += v * Wl[i];
        p2 += v * Wl[DHID + i];
    }
    #pragma unroll
    for (int off = 16; off > 0; off >>= 1) {
        p1 += __shfl_xor(p1, off, 32);
        p2 += __shfl_xor(p2, off, 32);
    }
    if (lane == 0) { s1[n] = p1; s2[n] = p2; }
}

__global__ void k_edge_pred(const float* __restrict__ s1, const float* __restrict__ s2,
                            const int* __restrict__ ei, const float* __restrict__ bl,
                            float* __restrict__ pred, int E) {
    int e = blockIdx.x * blockDim.x + threadIdx.x;
    if (e >= E) return;
    pred[e] = s1[ei[e]] + s2[ei[E + e]] + bl[0];
}

// per-graph bitonic sort (descending) of 512 edge scores, emit causal/conf splits
__global__ void k_sort_select(const float* __restrict__ pred, const int* __restrict__ ei,
                              float* __restrict__ out) {
    __shared__ float key[EPG];
    __shared__ int   idx[EPG];
    int b = blockIdx.x, t = threadIdx.x;
    key[t] = pred[b * EPG + t];
    idx[t] = t;
    __syncthreads();
    for (int k = 2; k <= EPG; k <<= 1) {
        for (int j = k >> 1; j > 0; j >>= 1) {
            int p = t ^ j;
            if (p > t) {
                bool desc = ((t & k) == 0);
                float kt = key[t], kp = key[p];
                bool sw = desc ? (kt < kp) : (kt > kp);
                if (sw) {
                    key[t] = kp; key[p] = kt;
                    int it = idx[t]; idx[t] = idx[p]; idx[p] = it;
                }
            }
            __syncthreads();
        }
    }
    const int HALF = NGRAPHS * NRES;                 // 65536
    float* causal_ei = out;                          // 2 x 65536
    float* causal_w  = out + 2 * HALF;               // 65536
    float* conf_ei   = out + 3 * HALF;               // 2 x 65536
    float* conf_w    = out + 5 * HALF;               // 65536
    int j = idx[t];
    float fsrc = (float)ei[0 * NEDGES + b * EPG + j];
    float fdst = (float)ei[1 * NEDGES + b * EPG + j];
    if (t < NRES) {
        causal_w[b * NRES + t] = key[t];
        causal_ei[0 * HALF + b * NRES + t] = fsrc;
        causal_ei[1 * HALF + b * NRES + t] = fdst;
    } else {
        int tt = t - NRES;
        conf_w[b * NRES + tt] = -key[t];
        conf_ei[0 * HALF + b * NRES + tt] = fsrc;
        conf_ei[1 * HALF + b * NRES + tt] = fdst;
    }
}

// ---------------------------------------------------------------- driver
extern "C" void kernel_launch(void* const* d_in, const int* in_sizes, int n_in,
                              void* d_out, int out_size, void* d_ws, size_t ws_size,
                              hipStream_t stream) {
    const float* x    = (const float*)d_in[0];
    const int*   ei   = (const int*)  d_in[1];
    const float* vemb = (const float*)d_in[3];
    const float* W1a  = (const float*)d_in[4];
    const float* b1a  = (const float*)d_in[5];
    const float* g1a  = (const float*)d_in[6];
    const float* be1a = (const float*)d_in[7];
    const float* W1b  = (const float*)d_in[8];
    const float* b1b  = (const float*)d_in[9];
    const float* g1   = (const float*)d_in[10];
    const float* be1  = (const float*)d_in[11];
    const float* W2a  = (const float*)d_in[12];
    const float* b2a  = (const float*)d_in[13];
    const float* g2a  = (const float*)d_in[14];
    const float* be2a = (const float*)d_in[15];
    const float* W2b  = (const float*)d_in[16];
    const float* b2b  = (const float*)d_in[17];
    const float* g2   = (const float*)d_in[18];
    const float* be2  = (const float*)d_in[19];
    const float* Wl   = (const float*)d_in[20];
    const float* bl   = (const float*)d_in[21];
    float* out = (float*)d_out;

    // workspace carve-up (256B aligned)
    char* ws = (char*)d_ws;
    size_t o = 0;
    auto take = [&](size_t bytes) { char* p = ws + o; o += (bytes + 255) & ~(size_t)255; return p; };
    _Float16* w1at  = (_Float16*)take((size_t)DMID * DIN  * 2);   // 512x128
    _Float16* w1bt  = (_Float16*)take((size_t)DHID * DMID * 2);   // 256x512
    _Float16* w2at  = (_Float16*)take((size_t)DMID * DHID * 2);   // 512x256
    _Float16* w2bt  = (_Float16*)take((size_t)DHID * DMID * 2);   // 256x512
    float*    stats = (float*)take(4 * DMID * 4);                 // sum|sumsq|mu|rsig
    float*    s1    = (float*)take((size_t)NNODES * 4);
    float*    s2    = (float*)take((size_t)NNODES * 4);
    float*    agg   = (float*)take((size_t)NNODES * DHID * 4);    // max-dim agg
    _Float16* hsum  = (_Float16*)take((size_t)NNODES * DHID * 2);
    float*    Z     = (float*)take((size_t)NNODES * DMID * 4);
    _Float16* Zf16  = (_Float16*)take((size_t)NNODES * DMID * 2);
    float*    H     = (float*)take((size_t)NNODES * DHID * 4);
    float*    post  = (float*)take((size_t)NNODES * DHID * 4);    // later reused as xn
    float* gsum = stats, *gsq = stats + DMID, *mu = stats + 2 * DMID, *rsig = stats + 3 * DMID;
    float* predO = out + 6 * NGRAPHS * NRES;                      // pred region of d_out

    const int TB = 256;
    auto gemm = [&](const _Float16* A, const _Float16* Bt, const float* bias, float* C,
                    int M, int Nn, int K) {
        int waves = (M >> 4) * (Nn >> 6);
        k_gemm_wmma<<<waves / 8, 256, 0, stream>>>(A, Bt, bias, C, M, Nn, K);
    };
    auto stats_pass = [&](const float* Zp, int C) {
        k_zero_f32<<<(2 * C + TB - 1) / TB, TB, 0, stream>>>(gsum, 2 * C);
        k_col_stats<<<NNODES / 128, TB, 0, stream>>>(Zp, gsum, gsq, C);
        k_finalize_stats<<<(C + TB - 1) / TB, TB, 0, stream>>>(gsum, gsq, mu, rsig, C, 1.f / NNODES);
    };

    // weight prep (tiny, once per call)
    k_transpose_f16<<<(DIN * DMID) / TB, TB, 0, stream>>>(W1a, w1at, DIN, DMID);
    k_transpose_f16<<<(DMID * DHID) / TB, TB, 0, stream>>>(W1b, w1bt, DMID, DHID);
    k_transpose_f16<<<(DHID * DMID) / TB, TB, 0, stream>>>(W2a, w2at, DHID, DMID);
    k_transpose_f16<<<(DMID * DHID) / TB, TB, 0, stream>>>(W2b, w2bt, DMID, DHID);

    // ---- GIN layer 1 ----
    k_zero_f32<<<(NNODES * DIN) / TB, TB, 0, stream>>>(agg, NNODES * DIN);
    k_scatter_add<<<(NEDGES * DIN) / TB, TB, 0, stream>>>(agg, x, ei, NEDGES, 7);
    k_sum_to_f16<<<(NNODES * DIN) / TB, TB, 0, stream>>>(x, agg, hsum, NNODES * DIN);
    gemm(hsum, w1at, b1a, Z, NNODES, DMID, DIN);
    stats_pass(Z, DMID);
    k_bn_relu_f16<<<(NNODES * DMID) / TB, TB, 0, stream>>>(Z, mu, rsig, g1a, be1a, Zf16,
                                                           NNODES * DMID, DMID - 1);
    gemm(Zf16, w1bt, b1b, H, NNODES, DHID, DMID);
    stats_pass(H, DHID);
    k_bn_relu_addv<<<(NNODES * DHID) / TB, TB, 0, stream>>>(H, mu, rsig, g1, be1, vemb, post,
                                                            NNODES * DHID, DHID - 1);

    // ---- GIN layer 2 ----
    k_zero_f32<<<(NNODES * DHID) / TB, TB, 0, stream>>>(agg, NNODES * DHID);
    k_scatter_add<<<(NEDGES * DHID) / TB, TB, 0, stream>>>(agg, post, ei, NEDGES, 8);
    k_sum_to_f16<<<(NNODES * DHID) / TB, TB, 0, stream>>>(post, agg, hsum, NNODES * DHID);
    gemm(hsum, w2at, b2a, Z, NNODES, DMID, DHID);
    stats_pass(Z, DMID);
    k_bn_relu_f16<<<(NNODES * DMID) / TB, TB, 0, stream>>>(Z, mu, rsig, g2a, be2a, Zf16,
                                                           NNODES * DMID, DMID - 1);
    gemm(Zf16, w2bt, b2b, H, NNODES, DHID, DMID);
    stats_pass(H, DHID);
    k_bn_f32<<<(NNODES * DHID) / TB, TB, 0, stream>>>(H, mu, rsig, g2, be2, post,
                                                      NNODES * DHID, DHID - 1);  // post := xn

    // ---- edge scoring + per-graph top-k split ----
    k_node_scores<<<NNODES / 8, 256, 0, stream>>>(post, Wl, s1, s2);
    k_edge_pred<<<NEDGES / TB, TB, 0, stream>>>(s1, s2, ei, bl, predO, NEDGES);
    k_sort_select<<<NGRAPHS, EPG, 0, stream>>>(predO, ei, out);

    (void)n_in; (void)in_sizes; (void)out_size; (void)ws_size;
}
